// glaudioLSTM_2241972929159
// MI455X (gfx1250) — matrix operand order repaired and checked by reading, compiled, not run on hardware
//
#include <hip/hip_runtime.h>
#include <cstdint>
#include <cstddef>

// ---------------------------------------------------------------------------
// GLAudio-style GNN+LSTM for MI455X (gfx1250, wave32, WMMA).
//  * All GEMMs via v_wmma_f32_16x16x32_bf16 (fp32 accumulate), K templated so
//    the K-loop fully unrolls into back-to-back WMMAs with b128 fragment loads.
//  * All GEMM A-operands pre-converted to bf16 once (no conversion ALU in the
//    WMMA inner loops).
//  * Message passing via CSR gather (no atomics in the 15-step wave loop).
//  * Backward LSTM = single cell step (reference uses hs_b[0]).
// ---------------------------------------------------------------------------

#define IN_DIM   128
#define HIDDEN   128
#define OUT_DIM  64
#define GATES    512
#define TSTEPS   16
#define HSTEP    0.1f

typedef __attribute__((ext_vector_type(16))) __bf16 v16bf;
typedef __attribute__((ext_vector_type(8)))  __bf16 v8bf;
typedef __attribute__((ext_vector_type(8)))  float  v8f;

__device__ __forceinline__ __bf16 f2bf(float f) {
  union { float f; unsigned u; } cv; cv.f = f;
  unsigned u = cv.u;
  u += 0x7FFFu + ((u >> 16) & 1u);            // round-to-nearest-even
  unsigned short s = (unsigned short)(u >> 16);
  __bf16 r; __builtin_memcpy(&r, &s, sizeof(r));
  return r;
}

// ---------------------------------------------------------------------------
// WMMA GEMM:  Out[M,N] = act( A[M,KT] @ W[N,KT]^T + bias )
//   ACC   : seed accumulator from existing OutF (read-modify-write GEMM)
//   ACT   : 0 = none, 1 = relu
//   OUTBF : write bf16 (OutB) instead of f32 (OutF)
//   KT    : compile-time K (row stride of A and W) -> fully unrolled K-loop
// One 16x16 output tile per wave; blockDim.x/32 waves tile N, blockIdx.x tiles M.
// All dims are exact multiples of the tile, EXEC stays all-ones (WMMA req).
// ---------------------------------------------------------------------------
template <bool ACC, int ACT, bool OUTBF, int KT>
__global__ void gemm_wmma(const __bf16* __restrict__ A,     // [M,KT] row-major
                          const __bf16* __restrict__ W,     // [N,KT] row-major
                          const float* __restrict__ bias,   // [N] or nullptr
                          float* __restrict__ OutF,
                          __bf16* __restrict__ OutB,
                          int M, int N) {
  const int wavesPerBlock = blockDim.x >> 5;
  const int wave = threadIdx.x >> 5;
  const int lane = threadIdx.x & 31;
  const int m0 = blockIdx.x * 16;
  const int n0 = (blockIdx.y * wavesPerBlock + wave) * 16;
  if (m0 >= M || n0 >= N) return;            // wave-uniform; never taken here

  const int r    = lane & 15;                // A row / B column within tile
  const int half = lane >> 4;

  v8f acc;
  if (ACC) {
#pragma unroll
    for (int i = 0; i < 8; ++i)
      acc[i] = OutF[(size_t)(m0 + half * 8 + i) * N + n0 + r];
  } else {
#pragma unroll
    for (int i = 0; i < 8; ++i) acc[i] = 0.0f;
  }

  const __bf16* arow = A + (size_t)(m0 + r) * KT;
  const __bf16* wrow = W + (size_t)(n0 + r) * KT;

#pragma unroll
  for (int k0 = 0; k0 < KT; k0 += 32) {
    // A 16x32 bf16 layout (ISA 7.12.2): lane<16 -> K 0..7 / 16..23 ;
    //                                   lane>=16 -> K 8..15 / 24..31
    const v8bf alo = *(const v8bf*)(arow + k0 + half * 8);        // b128
    const v8bf ahi = *(const v8bf*)(arow + k0 + 16 + half * 8);   // b128
    const v16bf af = __builtin_shufflevector(alo, ahi, 0, 1, 2, 3, 4, 5, 6, 7,
                                             8, 9, 10, 11, 12, 13, 14, 15);
    // B 32x16 bf16: lane = N column, same K order as A (K 0..7/16..23 vs 8..15/24..31)
    const v8bf blo = *(const v8bf*)(wrow + k0 + half * 8);        // b128
    const v8bf bhi = *(const v8bf*)(wrow + k0 + 16 + half * 8);   // b128
    const v16bf bf = __builtin_shufflevector(blo, bhi, 0, 1, 2, 3, 4, 5, 6, 7,
                                             8, 9, 10, 11, 12, 13, 14, 15);

    acc = __builtin_amdgcn_wmma_f32_16x16x32_bf16(
        /*neg_a=*/false, af, /*neg_b=*/false, bf,
        /*c_mod=*/(short)0, acc, /*reuse_a=*/false, /*reuse_b=*/false);
  }

  const float bv = bias ? bias[n0 + r] : 0.0f;
#pragma unroll
  for (int i = 0; i < 8; ++i) {
    const int m = m0 + half * 8 + i;
    float v = acc[i] + bv;
    if (ACT == 1) v = fmaxf(v, 0.0f);
    if (OUTBF) OutB[(size_t)m * N + n0 + r] = f2bf(v);
    else       OutF[(size_t)m * N + n0 + r] = v;
  }
}

// ---------------------------------------------------------------------------
// Pointwise / graph kernels
// ---------------------------------------------------------------------------
__global__ void cvt_f32_bf16(const float* __restrict__ in,
                             __bf16* __restrict__ out, size_t n) {
  size_t i = (size_t)blockIdx.x * blockDim.x + threadIdx.x;
  if (i < n) out[i] = f2bf(in[i]);
}

__global__ void add_vec(const float* a, const float* b, float* out, int n) {
  int i = blockIdx.x * blockDim.x + threadIdx.x;
  if (i < n) out[i] = a[i] + b[i];
}

__global__ void degree_count(const int* __restrict__ col, int* __restrict__ counts,
                             int nE) {
  int e = blockIdx.x * blockDim.x + threadIdx.x;
  if (e < nE) atomicAdd(&counts[col[e]], 1);
}

// Exclusive scan of counts[0..n) into offsets[0..n]; single block of 1024.
__global__ void exscan_1024(const int* __restrict__ counts,
                            int* __restrict__ offsets, int n) {
  __shared__ int buf[1024];
  __shared__ int carry_s;
  if (threadIdx.x == 0) carry_s = 0;
  __syncthreads();
  for (int base = 0; base < n; base += 1024) {
    int i = base + (int)threadIdx.x;
    int v = (i < n) ? counts[i] : 0;
    buf[threadIdx.x] = v;
    __syncthreads();
    for (int off = 1; off < 1024; off <<= 1) {
      int t = (threadIdx.x >= (unsigned)off) ? buf[threadIdx.x - off] : 0;
      __syncthreads();
      buf[threadIdx.x] += t;
      __syncthreads();
    }
    int carry = carry_s;
    if (i < n) offsets[i] = carry + buf[threadIdx.x] - v;
    __syncthreads();
    if (threadIdx.x == 1023) carry_s = carry + buf[1023];
    __syncthreads();
  }
  if (threadIdx.x == 0) offsets[n] = carry_s;
}

__global__ void csr_fill(const int* __restrict__ row, const int* __restrict__ col,
                         const int* __restrict__ offsets, int* __restrict__ cursor,
                         int* __restrict__ esrc, int nE) {
  int e = blockIdx.x * blockDim.x + threadIdx.x;
  if (e < nE) {
    int c = col[e];
    int pos = atomicAdd(&cursor[c], 1);
    esrc[offsets[c] + pos] = row[e];
  }
}

// One block per node, one thread per feature dim.
//   mes_i = -sum_{j->i} X_j ;  vel = deg_i*X_i + mes ;  Y -= H*vel ;  X += H*Y
__global__ void wave_step(const float* __restrict__ Xc, float* __restrict__ Y,
                          float* __restrict__ Xn, __bf16* __restrict__ xt_out,
                          const int* __restrict__ start,
                          const int* __restrict__ esrc) {
  const int node = blockIdx.x;
  const int d = threadIdx.x;
  const int s = start[node], e = start[node + 1];
  float acc = 0.0f;
  for (int i = s; i < e; ++i) {
    if (i + 1 < e)   // emits global_prefetch_b8 on gfx1250
      __builtin_prefetch(&Xc[(size_t)esrc[i + 1] * HIDDEN + d], 0, 1);
    acc += Xc[(size_t)esrc[i] * HIDDEN + d];
  }
  const size_t idx = (size_t)node * HIDDEN + d;
  const float xc  = Xc[idx];
  const float deg = (float)(e - s);
  const float vel = deg * xc - acc;
  const float yn  = Y[idx] - HSTEP * vel;
  const float xn  = xc + HSTEP * yn;
  Y[idx] = yn;
  Xn[idx] = xn;
  xt_out[idx] = f2bf(xn);
}

// LSTM cell pointwise part; gates layout (torch order): [i | f | g | o]
__global__ void lstm_cell(const float* __restrict__ gates, float* __restrict__ h,
                          float* __restrict__ c, __bf16* __restrict__ hbf,
                          int nNodes) {
  const int idx = blockIdx.x * blockDim.x + threadIdx.x;
  if (idx >= nNodes * HIDDEN) return;
  const int node = idx >> 7, d = idx & 127;
  const size_t g0 = (size_t)node * GATES + d;
  const float ig = 1.0f / (1.0f + __expf(-gates[g0]));
  const float fg = 1.0f / (1.0f + __expf(-gates[g0 + 128]));
  const float gg = tanhf(gates[g0 + 256]);
  const float og = 1.0f / (1.0f + __expf(-gates[g0 + 384]));
  const float cn = fg * c[idx] + ig * gg;
  const float hn = og * tanhf(cn);
  c[idx] = cn;
  h[idx] = hn;
  if (hbf) hbf[idx] = f2bf(hn);
}

__global__ void concat_bf16(const float* __restrict__ hf,
                            const float* __restrict__ hb,
                            __bf16* __restrict__ out, int nNodes) {
  const int idx = blockIdx.x * blockDim.x + threadIdx.x;
  if (idx >= nNodes * HIDDEN) return;
  const int node = idx >> 7, d = idx & 127;
  out[(size_t)node * (2 * HIDDEN) + d]          = f2bf(hf[idx]);
  out[(size_t)node * (2 * HIDDEN) + HIDDEN + d] = f2bf(hb[idx]);
}

// ---------------------------------------------------------------------------
// Host orchestration
// ---------------------------------------------------------------------------
extern "C" void kernel_launch(void* const* d_in, const int* in_sizes, int n_in,
                              void* d_out, int out_size, void* d_ws, size_t ws_size,
                              hipStream_t stream) {
  (void)n_in; (void)out_size; (void)ws_size;

  const float* x      = (const float*)d_in[0];
  const int*   eidx   = (const int*)d_in[1];    // [2, E] : row then col
  const float* W_enc1 = (const float*)d_in[2];
  const float* b_enc1 = (const float*)d_in[3];
  const float* W_enc2 = (const float*)d_in[4];
  const float* b_enc2 = (const float*)d_in[5];
  const float* W_ih_f = (const float*)d_in[6];
  const float* W_hh_f = (const float*)d_in[7];
  const float* b_ih_f = (const float*)d_in[8];
  const float* b_hh_f = (const float*)d_in[9];
  const float* W_ih_b = (const float*)d_in[10];
  // d_in[11] (W_hh_b) unused: backward scan only contributes hs_b[0] with h0=0
  const float* b_ih_b = (const float*)d_in[12];
  const float* b_hh_b = (const float*)d_in[13];
  const float* W_f1   = (const float*)d_in[14];
  const float* b_f1   = (const float*)d_in[15];
  const float* W_f2   = (const float*)d_in[16];
  const float* b_f2   = (const float*)d_in[17];
  float* out = (float*)d_out;

  const int    NN = in_sizes[0] / IN_DIM;   // 20000 (multiple of 16)
  const int    NE = in_sizes[1] / 2;        // 640000
  const size_t NF = (size_t)NN * HIDDEN;

  // ---- workspace bump allocator (256B aligned) ----
  char* p = (char*)d_ws;
  auto bump = [&](size_t bytes) -> char* {
    char* r = p;
    p += (bytes + 255) & ~(size_t)255;
    return r;
  };
  __bf16* wbE1  = (__bf16*)bump((size_t)HIDDEN * IN_DIM * 2);
  __bf16* wbE2  = (__bf16*)bump((size_t)HIDDEN * HIDDEN * 2);
  __bf16* wbIHf = (__bf16*)bump((size_t)GATES * HIDDEN * 2);
  __bf16* wbHHf = (__bf16*)bump((size_t)GATES * HIDDEN * 2);
  __bf16* wbIHb = (__bf16*)bump((size_t)GATES * HIDDEN * 2);
  __bf16* wbF1  = (__bf16*)bump((size_t)HIDDEN * 2 * HIDDEN * 2);
  __bf16* wbF2  = (__bf16*)bump((size_t)OUT_DIM * HIDDEN * 2);
  float*  biasF = (float*)bump(GATES * 4);
  float*  biasB = (float*)bump(GATES * 4);
  __bf16* xbf   = (__bf16*)bump((size_t)NN * IN_DIM * 2);  // x as bf16
  __bf16* Xb1   = (__bf16*)bump(NF * 2);                   // enc1 activation
  float*  Xa    = (float*)bump(NF * 4);                    // wave ping
  float*  Xpb   = (float*)bump(NF * 4);                    // wave pong
  float*  Yb    = (float*)bump(NF * 4);
  __bf16* Xt    = (__bf16*)bump((size_t)TSTEPS * NF * 2);  // X_tilde, bf16
  int*    counts = (int*)bump((size_t)NN * 4);
  int*    offs   = (int*)bump((size_t)(NN + 1) * 4);
  int*    cursor = (int*)bump((size_t)NN * 4);
  int*    esrc   = (int*)bump((size_t)NE * 4);
  float*  gates  = (float*)bump((size_t)NN * GATES * 4);
  float*  hf     = (float*)bump(NF * 4);
  float*  cfb    = (float*)bump(NF * 4);
  __bf16* hbf    = (__bf16*)bump(NF * 2);
  float*  hb     = (float*)bump(NF * 4);
  float*  cbb    = (float*)bump(NF * 4);
  __bf16* catb   = (__bf16*)bump((size_t)NN * 2 * HIDDEN * 2);
  __bf16* hid2   = (__bf16*)bump(NF * 2);

  // ---- zero-init (capture-safe) ----
  hipMemsetAsync(Yb,     0, NF * 4, stream);
  hipMemsetAsync(counts, 0, (size_t)NN * 4, stream);
  hipMemsetAsync(cursor, 0, (size_t)NN * 4, stream);
  hipMemsetAsync(hf,     0, NF * 4, stream);
  hipMemsetAsync(cfb,    0, NF * 4, stream);
  hipMemsetAsync(hb,     0, NF * 4, stream);
  hipMemsetAsync(cbb,    0, NF * 4, stream);

  // ---- f32 -> bf16 conversions (weights + input activations) ----
  auto cvt = [&](const float* src, __bf16* dst, size_t n) {
    cvt_f32_bf16<<<(unsigned)((n + 255) / 256), 256, 0, stream>>>(src, dst, n);
  };
  cvt(W_enc1, wbE1,  (size_t)HIDDEN * IN_DIM);
  cvt(W_enc2, wbE2,  (size_t)HIDDEN * HIDDEN);
  cvt(W_ih_f, wbIHf, (size_t)GATES * HIDDEN);
  cvt(W_hh_f, wbHHf, (size_t)GATES * HIDDEN);
  cvt(W_ih_b, wbIHb, (size_t)GATES * HIDDEN);
  cvt(W_f1,   wbF1,  (size_t)HIDDEN * 2 * HIDDEN);
  cvt(W_f2,   wbF2,  (size_t)OUT_DIM * HIDDEN);
  cvt(x,      xbf,   (size_t)NN * IN_DIM);
  add_vec<<<2, 256, 0, stream>>>(b_ih_f, b_hh_f, biasF, GATES);
  add_vec<<<2, 256, 0, stream>>>(b_ih_b, b_hh_b, biasB, GATES);

  const int MT = NN / 16;  // 1250 M-tiles

  // ---- encoders ----
  {  // X = relu(x @ W_enc1^T + b_enc1)  -> bf16
    dim3 g(MT, HIDDEN / (16 * 8));
    gemm_wmma<false, 1, true, IN_DIM><<<g, 256, 0, stream>>>(
        xbf, wbE1, b_enc1, nullptr, Xb1, NN, HIDDEN);
  }
  {  // X = X @ W_enc2^T + b_enc2  -> f32 (wave state)
    dim3 g(MT, HIDDEN / (16 * 8));
    gemm_wmma<false, 0, false, HIDDEN><<<g, 256, 0, stream>>>(
        Xb1, wbE2, b_enc2, Xa, nullptr, NN, HIDDEN);
  }
  cvt_f32_bf16<<<(unsigned)((NF + 255) / 256), 256, 0, stream>>>(Xa, Xt, NF);

  // ---- CSR build (once): count -> scan -> fill ----
  degree_count<<<(NE + 255) / 256, 256, 0, stream>>>(eidx + NE, counts, NE);
  exscan_1024<<<1, 1024, 0, stream>>>(counts, offs, NN);
  csr_fill<<<(NE + 255) / 256, 256, 0, stream>>>(eidx, eidx + NE, offs, cursor,
                                                 esrc, NE);

  // ---- 15 wave steps (gather-only, X fits in L2) ----
  float* curX = Xa;
  float* nxtX = Xpb;
  for (int t = 1; t < TSTEPS; ++t) {
    wave_step<<<NN, HIDDEN, 0, stream>>>(curX, Yb, nxtX, Xt + (size_t)t * NF,
                                         offs, esrc);
    float* tmp = curX; curX = nxtX; nxtX = tmp;
  }

  // ---- forward LSTM scan ----
  dim3 gg(MT, GATES / (16 * 8));  // (1250, 4)
  for (int t = 0; t < TSTEPS; ++t) {
    gemm_wmma<false, 0, false, HIDDEN><<<gg, 256, 0, stream>>>(
        Xt + (size_t)t * NF, wbIHf, biasF, gates, nullptr, NN, GATES);
    if (t > 0)  // h == 0 at t==0
      gemm_wmma<true, 0, false, HIDDEN><<<gg, 256, 0, stream>>>(
          hbf, wbHHf, nullptr, gates, nullptr, NN, GATES);
    lstm_cell<<<(unsigned)((NF + 255) / 256), 256, 0, stream>>>(gates, hf, cfb,
                                                                hbf, NN);
  }

  // ---- backward LSTM: reference uses hs_b[0] == single cell on X_tilde[T-1]
  gemm_wmma<false, 0, false, HIDDEN><<<gg, 256, 0, stream>>>(
      Xt + (size_t)(TSTEPS - 1) * NF, wbIHb, biasB, gates, nullptr, NN, GATES);
  lstm_cell<<<(unsigned)((NF + 255) / 256), 256, 0, stream>>>(gates, hb, cbb,
                                                              nullptr, NN);

  // ---- head ----
  concat_bf16<<<(unsigned)((NF + 255) / 256), 256, 0, stream>>>(hf, hb, catb, NN);
  {  // relu(cat @ W_f1^T + b_f1) -> bf16
    dim3 g(MT, HIDDEN / (16 * 8));
    gemm_wmma<false, 1, true, 2 * HIDDEN><<<g, 256, 0, stream>>>(
        catb, wbF1, b_f1, nullptr, hid2, NN, HIDDEN);
  }
  {  // hid2 @ W_f2^T + b_f2 -> f32 output (N=64 -> 4 waves / 128 threads)
    dim3 g(MT, OUT_DIM / (16 * 4));
    gemm_wmma<false, 0, false, HIDDEN><<<g, 128, 0, stream>>>(
        hid2, wbF2, b_f2, out, nullptr, NN, OUT_DIM);
  }
}
